// PyTorchLSTM_23845658427405
// MI455X (gfx1250) — compile-verified, matrix operand order checked by reading, not run on hardware
//
#include <hip/hip_runtime.h>

// ---------------------------------------------------------------------------
// Problem constants (match reference)
// ---------------------------------------------------------------------------
#define BSZ   256
#define TLEN  256
#define EMBD  512
#define HID   1024
#define GATES 4096   // 4*HID

// Step-kernel tiling: block = 32 batch x 64 hidden, 8 waves, 2x2 reg blocking.
#define MT    32     // batch tile per block
#define NT    64     // hidden tile per block
#define LDS_STRIDE 68  // padded row stride (floats) to break bank aliasing

typedef __attribute__((ext_vector_type(16))) __bf16 v16bf;
typedef __attribute__((ext_vector_type(8)))  __bf16 v8bf;
typedef __attribute__((ext_vector_type(8)))  float  v8f;

// ---------------------------------------------------------------------------
// Helpers
// ---------------------------------------------------------------------------
__device__ __forceinline__ unsigned short f32_to_bf16(float f) {
    union { float f; unsigned int u; } v; v.f = f;
    unsigned int u = v.u;
    unsigned int r = u + 0x7FFFu + ((u >> 16) & 1u);   // round-to-nearest-even
    if ((u & 0x7F800000u) == 0x7F800000u) r = u;       // inf/nan passthrough
    return (unsigned short)(r >> 16);
}

// Load one 16x32 bf16 WMMA fragment element set for this lane.
// p points at (row_base + khalf) where khalf = (lane>=16 ? 8 : 0).
// halves [0..7]  = K(khalf+0 .. khalf+7)
// halves [8..15] = K(khalf+16.. khalf+23)   (per CDNA5 ISA 16-bit A layout)
__device__ __forceinline__ v16bf load_frag(const unsigned short* p) {
    v8bf lo = *reinterpret_cast<const v8bf*>(p);
    v8bf hi = *reinterpret_cast<const v8bf*>(p + 16);
    return __builtin_shufflevector(lo, hi, 0, 1, 2, 3, 4, 5, 6, 7,
                                           8, 9, 10, 11, 12, 13, 14, 15);
}

__device__ __forceinline__ float sigmoidf_fast(float x) {
    return 1.0f / (1.0f + __expf(-x));
}

// ---------------------------------------------------------------------------
// Kernel: f32 -> bf16 bulk convert (weights)
// ---------------------------------------------------------------------------
__global__ void f32_to_bf16_kernel(const float* __restrict__ in,
                                   unsigned short* __restrict__ out,
                                   unsigned int n) {
    unsigned int i = blockIdx.x * blockDim.x + threadIdx.x;
    if (i < n) out[i] = f32_to_bf16(in[i]);
}

// ---------------------------------------------------------------------------
// Kernel: fused bias  (b_ih + b_hh)
// ---------------------------------------------------------------------------
__global__ void bias_kernel(const float* __restrict__ bih,
                            const float* __restrict__ bhh,
                            float* __restrict__ bias, int n) {
    int i = blockIdx.x * blockDim.x + threadIdx.x;
    if (i < n) bias[i] = bih[i] + bhh[i];
}

// ---------------------------------------------------------------------------
// Kernel: embedding gather + f32->bf16   x[b,t,:] = bf16(emb[text[b,t],:])
// ---------------------------------------------------------------------------
__global__ void gather_embed_kernel(const int* __restrict__ text,
                                    const float* __restrict__ emb,
                                    unsigned short* __restrict__ xbf) {
    const int row = blockIdx.x;                 // b*T + t
    const int idx = text[row];
    const float* src = emb + (size_t)idx * EMBD;
    unsigned short* dst = xbf + (size_t)row * EMBD;
    for (int e = threadIdx.x; e < EMBD; e += blockDim.x)
        dst[e] = f32_to_bf16(src[e]);
}

// ---------------------------------------------------------------------------
// Kernel: zero recurrent state
// ---------------------------------------------------------------------------
__global__ void zero_state_kernel(float* __restrict__ c,
                                  unsigned short* __restrict__ h0,
                                  unsigned short* __restrict__ h1) {
    int i = blockIdx.x * blockDim.x + threadIdx.x;
    if (i < BSZ * HID) { c[i] = 0.0f; h0[i] = 0; h1[i] = 0; }
}

// ---------------------------------------------------------------------------
// Kernel: one LSTM timestep.
//   gates = [x_t | h_in] * [w_ih | w_hh]^T + (b_ih + b_hh)   K = 1536, bf16 WMMA
//   c = sig(f)*c + sig(i)*tanh(g);  h = sig(o)*tanh(c)
//
// Block: 256 thr = 8 waves, tile = 32 batch x 64 hidden.
// Wave w -> gate (w>>1), n-pair (w&1); holds 2 A-frags x 2 B-frags ->
// 4 independent accumulators (8 b128 loads feed 4 WMMAs = 32 B/WMMA).
// Grid: (B/32, H/64) = (8, 16).
// ---------------------------------------------------------------------------
__global__ __launch_bounds__(256) void lstm_step_kernel(
    const unsigned short* __restrict__ xbf,   // [B, T, EMBD] bf16
    const unsigned short* __restrict__ wih,   // [GATES, EMBD] bf16
    const unsigned short* __restrict__ whh,   // [GATES, HID]  bf16
    const float*          __restrict__ bias,  // [GATES]
    const unsigned short* __restrict__ h_in,  // [B, HID] bf16
    unsigned short*       __restrict__ h_out, // [B, HID] bf16
    float*                __restrict__ h_f32, // [B, HID]
    float*                __restrict__ c,     // [B, HID]
    int t)
{
    __shared__ float gl[4 * MT * LDS_STRIDE];   // 4 gates x 32 x 68 f32 = 34 KB

    const int tid   = threadIdx.x;
    const int lane  = tid & 31;
    const int wave  = tid >> 5;
    const int gate  = wave >> 1;              // 0..3  (i,f,g,o)
    const int npair = wave & 1;               // 0..1  (which 32-col half)
    const int l16   = lane & 15;
    const int lhi   = lane >> 4;              // 0/1
    const int khalf = lhi * 8;                // K shift for this lane half

    const int m_base = blockIdx.x * MT;
    const int m0     = m_base + l16;                       // A rows (batch)
    const int n_hid0 = blockIdx.y * NT + npair * 32 + l16; // B cols (hidden)
    const int gcol0  = gate * HID + n_hid0;                // weight rows
    const int gcol1  = gcol0 + 16;

    // Accumulators pre-loaded with bias (column-constant across each tile).
    const float bv0 = bias[gcol0];
    const float bv1 = bias[gcol1];
    v8f acc00 = { bv0, bv0, bv0, bv0, bv0, bv0, bv0, bv0 };
    v8f acc01 = { bv1, bv1, bv1, bv1, bv1, bv1, bv1, bv1 };
    v8f acc10 = acc00;
    v8f acc11 = acc01;

    // ---- K part 1: x_t contribution (K = 512) ----
    {
        const unsigned short* a0p = xbf + ((size_t)m0 * TLEN + t) * EMBD + khalf;
        const unsigned short* a1p = a0p + (size_t)16 * TLEN * EMBD;
        const unsigned short* b0p = wih + (size_t)gcol0 * EMBD + khalf;
        const unsigned short* b1p = wih + (size_t)gcol1 * EMBD + khalf;
        #pragma unroll 2
        for (int k0 = 0; k0 < EMBD; k0 += 32) {
            v16bf a0 = load_frag(a0p + k0);
            v16bf a1 = load_frag(a1p + k0);
            v16bf b0 = load_frag(b0p + k0);
            v16bf b1 = load_frag(b1p + k0);
            acc00 = __builtin_amdgcn_wmma_f32_16x16x32_bf16(
                        false, a0, false, b0, (short)0, acc00, false, false);
            acc01 = __builtin_amdgcn_wmma_f32_16x16x32_bf16(
                        false, a0, false, b1, (short)0, acc01, false, false);
            acc10 = __builtin_amdgcn_wmma_f32_16x16x32_bf16(
                        false, a1, false, b0, (short)0, acc10, false, false);
            acc11 = __builtin_amdgcn_wmma_f32_16x16x32_bf16(
                        false, a1, false, b1, (short)0, acc11, false, false);
        }
    }

    // ---- K part 2: h contribution (K = 1024) ----
    {
        const unsigned short* a0p = h_in + (size_t)m0 * HID + khalf;
        const unsigned short* a1p = a0p + (size_t)16 * HID;
        const unsigned short* b0p = whh + (size_t)gcol0 * HID + khalf;
        const unsigned short* b1p = whh + (size_t)gcol1 * HID + khalf;
        #pragma unroll 2
        for (int k0 = 0; k0 < HID; k0 += 32) {
            v16bf a0 = load_frag(a0p + k0);
            v16bf a1 = load_frag(a1p + k0);
            v16bf b0 = load_frag(b0p + k0);
            v16bf b1 = load_frag(b1p + k0);
            acc00 = __builtin_amdgcn_wmma_f32_16x16x32_bf16(
                        false, a0, false, b0, (short)0, acc00, false, false);
            acc01 = __builtin_amdgcn_wmma_f32_16x16x32_bf16(
                        false, a0, false, b1, (short)0, acc01, false, false);
            acc10 = __builtin_amdgcn_wmma_f32_16x16x32_bf16(
                        false, a1, false, b0, (short)0, acc10, false, false);
            acc11 = __builtin_amdgcn_wmma_f32_16x16x32_bf16(
                        false, a1, false, b1, (short)0, acc11, false, false);
        }
    }

    // ---- Spill 4 D-tiles to LDS (D layout: m = vgpr + 8*lhi, n = l16) ----
    {
        float* gbase = gl + gate * (MT * LDS_STRIDE);
        const int ncol0 = npair * 32 + l16;
        #pragma unroll
        for (int v = 0; v < 8; ++v) {
            const int mm = v + 8 * lhi;
            gbase[(mm     ) * LDS_STRIDE + ncol0     ] = acc00[v];
            gbase[(mm     ) * LDS_STRIDE + ncol0 + 16] = acc01[v];
            gbase[(mm + 16) * LDS_STRIDE + ncol0     ] = acc10[v];
            gbase[(mm + 16) * LDS_STRIDE + ncol0 + 16] = acc11[v];
        }
    }
    __syncthreads();

    // ---- Elementwise gate recombination + state update (32*64 elems) ----
    for (int e = tid; e < MT * NT; e += 256) {
        const int mloc = e >> 6;     // 0..31
        const int j    = e & 63;     // 0..63 hidden within tile
        const int base = mloc * LDS_STRIDE + j;

        const float gi = gl[0 * (MT * LDS_STRIDE) + base];
        const float gf = gl[1 * (MT * LDS_STRIDE) + base];
        const float gg = gl[2 * (MT * LDS_STRIDE) + base];
        const float go = gl[3 * (MT * LDS_STRIDE) + base];

        const int mg = m_base + mloc;
        const int jg = blockIdx.y * NT + j;
        const size_t idx = (size_t)mg * HID + jg;

        const float cv = c[idx];
        const float cn = sigmoidf_fast(gf) * cv + sigmoidf_fast(gi) * tanhf(gg);
        c[idx] = cn;
        const float hn = sigmoidf_fast(go) * tanhf(cn);
        h_out[idx] = f32_to_bf16(hn);
        h_f32[idx] = hn;
    }
}

// ---------------------------------------------------------------------------
// Kernel: final FC + sigmoid.  out[b] = sig(h[b,:] . fc_w + fc_b)
// ---------------------------------------------------------------------------
__global__ __launch_bounds__(256) void fc_sigmoid_kernel(
    const float* __restrict__ h,      // [B, HID] f32
    const float* __restrict__ fc_w,   // [HID]
    const float* __restrict__ fc_b,   // [1]
    float* __restrict__ out)          // [B]
{
    __shared__ float red[256];
    const int b = blockIdx.x;
    float s = 0.0f;
    for (int k = threadIdx.x; k < HID; k += 256)
        s += h[(size_t)b * HID + k] * fc_w[k];
    red[threadIdx.x] = s;
    __syncthreads();
    for (int off = 128; off > 0; off >>= 1) {
        if (threadIdx.x < off) red[threadIdx.x] += red[threadIdx.x + off];
        __syncthreads();
    }
    if (threadIdx.x == 0)
        out[b] = sigmoidf_fast(red[0] + fc_b[0]);
}

// ---------------------------------------------------------------------------
// Launch
// ---------------------------------------------------------------------------
extern "C" void kernel_launch(void* const* d_in, const int* in_sizes, int n_in,
                              void* d_out, int out_size, void* d_ws, size_t ws_size,
                              hipStream_t stream) {
    const int*   text = (const int*)  d_in[0];
    const float* emb  = (const float*)d_in[1];
    const float* w_ih = (const float*)d_in[2];
    const float* w_hh = (const float*)d_in[3];
    const float* b_ih = (const float*)d_in[4];
    const float* b_hh = (const float*)d_in[5];
    const float* fc_w = (const float*)d_in[6];
    const float* fc_b = (const float*)d_in[7];
    float*       out  = (float*)d_out;

    // Workspace carve-up (~79 MB total)
    char* ws = (char*)d_ws;
    size_t off = 0;
    auto carve = [&](size_t bytes) -> void* {
        void* p = ws + off;
        off += (bytes + 255) & ~(size_t)255;
        return p;
    };
    unsigned short* wih_bf = (unsigned short*)carve((size_t)GATES * EMBD * 2); //  4 MB
    unsigned short* whh_bf = (unsigned short*)carve((size_t)GATES * HID  * 2); //  8 MB
    float*          bias   = (float*)         carve((size_t)GATES * 4);        // 16 KB
    unsigned short* xbf    = (unsigned short*)carve((size_t)BSZ * TLEN * EMBD * 2); // 64 MB
    unsigned short* h0     = (unsigned short*)carve((size_t)BSZ * HID * 2);    // 512 KB
    unsigned short* h1     = (unsigned short*)carve((size_t)BSZ * HID * 2);    // 512 KB
    float*          hf32   = (float*)         carve((size_t)BSZ * HID * 4);    //   1 MB
    float*          cbuf   = (float*)         carve((size_t)BSZ * HID * 4);    //   1 MB

    // Phase 0: weight/bias conversion + embedding gather + state init
    f32_to_bf16_kernel<<<(GATES * EMBD + 255) / 256, 256, 0, stream>>>(
        w_ih, wih_bf, GATES * EMBD);
    f32_to_bf16_kernel<<<(GATES * HID + 255) / 256, 256, 0, stream>>>(
        w_hh, whh_bf, GATES * HID);
    bias_kernel<<<(GATES + 255) / 256, 256, 0, stream>>>(b_ih, b_hh, bias, GATES);
    gather_embed_kernel<<<BSZ * TLEN, 256, 0, stream>>>(text, emb, xbf);
    zero_state_kernel<<<(BSZ * HID + 255) / 256, 256, 0, stream>>>(cbuf, h0, h1);

    // Phase 1: sequential recurrence, one launch per timestep (graph-friendly).
    // h is double-buffered: read parity t, write parity t+1.
    for (int t = 0; t < TLEN; ++t) {
        unsigned short* hin  = (t & 1) ? h1 : h0;
        unsigned short* hout = (t & 1) ? h0 : h1;
        lstm_step_kernel<<<dim3(BSZ / MT, HID / NT), 256, 0, stream>>>(
            xbf, wih_bf, whh_bf, bias, hin, hout, hf32, cbuf, t);
    }

    // Phase 2: final projection + sigmoid
    fc_sigmoid_kernel<<<BSZ, 256, 0, stream>>>(hf32, fc_w, fc_b, out);
}
